// Attention_23433341567267
// MI455X (gfx1250) — matrix operand order repaired and checked by reading, not compiled, not run on hardware
//
#include <hip/hip_runtime.h>

// ---------------------------------------------------------------------------
// Masked multi-softmax attention for MI455X (gfx1250, wave32, WMMA).
// B=8 H=12 S=512 D=64.  Outputs (concatenated f32):
//   out[B,H,S,D], p_attn, p_ww, p_wp, p_pw, p_pp  (each [B,H,S,S])
// One block = one (b, h, 16-row query tile); 5 waves, one score matrix each.
// Two-pass (recompute) softmax in base-2 (exp2 = HW op). Pass 1 keeps
// PER-LANE online max/sum and a per-lane mask bitmask; one 16-lane combine
// at the end. Pass 2 recomputes scores (inputs are L2-resident: ~96MB vs
// 192MB L2), normalizes, and streams probs with non-temporal stores so the
// 516MB output stream does not thrash L2. Wave 0 also does out = P*V.
// ---------------------------------------------------------------------------

typedef __attribute__((ext_vector_type(16))) __bf16 v16bf;
typedef __attribute__((ext_vector_type(8)))  float  v8f;

#define LOG2E    (1.44269504088896340736f)
#define NEG_INF2 (-1000000000.0f * LOG2E)   // mask value in base-2 domain
#define SCALE2   (0.125f * LOG2E)           // (1/sqrt(64)) * log2(e)

__device__ __forceinline__ float red_max16(float v) {
  v = fmaxf(v, __shfl_xor(v, 1, 32));
  v = fmaxf(v, __shfl_xor(v, 2, 32));
  v = fmaxf(v, __shfl_xor(v, 4, 32));
  v = fmaxf(v, __shfl_xor(v, 8, 32));
  return v;                        // max over each 16-lane half
}
__device__ __forceinline__ float red_sum16(float v) {
  v += __shfl_xor(v, 1, 32);
  v += __shfl_xor(v, 2, 32);
  v += __shfl_xor(v, 4, 32);
  v += __shfl_xor(v, 8, 32);
  return v;                        // sum over each 16-lane half
}

__global__ __launch_bounds__(160)
void attn_multi_softmax_kernel(const float* __restrict__ q,
                               const float* __restrict__ kk,
                               const float* __restrict__ vv,
                               const int*   __restrict__ mask,
                               const float* __restrict__ o1,
                               const float* __restrict__ o2,
                               const float* __restrict__ p1,
                               const float* __restrict__ p2,
                               float* __restrict__ out)
{
  constexpr int H = 12, S = 512, Dh = 64, Bn = 8;
  const int qt   = blockIdx.x;           // query tile (16 rows)
  const int h    = blockIdx.y;
  const int b    = blockIdx.z;
  const int w    = threadIdx.x >> 5;     // wave id 0..4 -> which score matrix
  const int lane = threadIdx.x & 31;
  const int row  = lane & 15;            // A-fragment row / C-tile column
  const int half = lane >> 4;

  __shared__ float stage[16 * 32];       // P-tile pair staging (wave 0 only)

  const float* Asel[5] = { q,  o1, o1, p1, p1 };
  const float* Bsel[5] = { kk, o2, p2, o2, p2 };

  const size_t bh    = (size_t)(b * H + h);
  const float* Abase = Asel[w] + (bh * S + (size_t)qt * 16) * Dh;   // [16,64]
  const float* Bbase = Bsel[w] + bh * S * Dh;                       // [512,64]
  const float* Vbase = vv      + bh * S * Dh;                       // [512,64]
  const int*   mbase = mask + ((size_t)b * S + (size_t)qt * 16) * S;

  const size_t OUT0 = (size_t)Bn * H * S * Dh;
  const size_t BHSS = (size_t)Bn * H * S * S;
  float* probs = out + OUT0 + (size_t)w * BHSS + bh * S * S + (size_t)qt * 16 * S;
  float* obase = out + (bh * S + (size_t)qt * 16) * Dh;

  // ---- load A fragments (16x64 bf16 -> two K=32 fragments), f32->bf16 ----
  v16bf a0, a1;
#pragma unroll
  for (int p = 0; p < 8; ++p) {
    const int ke = half * 16 + p * 2;                       // A layout K index
    float2 t0 = *(const float2*)(Abase + row * Dh + ke);
    float2 t1 = *(const float2*)(Abase + row * Dh + 32 + ke);
    a0[2 * p] = (__bf16)t0.x;  a0[2 * p + 1] = (__bf16)t0.y;
    a1[2 * p] = (__bf16)t1.x;  a1[2 * p + 1] = (__bf16)t1.y;
  }

  // per-LANE online softmax state (each lane owns 32 columns of each row)
  float lmax[8], lsum[8];
  unsigned mbits[8];               // mask bit n -> tile n valid for this row
#pragma unroll
  for (int r = 0; r < 8; ++r) { lmax[r] = -1.0e30f; lsum[r] = 0.0f; mbits[r] = 0u; }

  // ================= pass 1: per-lane online row max / sum =================
  for (int n = 0; n < 32; ++n) {
    const float* Bt = Bbase + (size_t)(n * 16 + row) * Dh;  // col of B = row of K
    if (n + 1 < 32)
      __builtin_prefetch(Bbase + (size_t)((n + 1) * 16 + row) * Dh, 0, 0);
    v16bf b0, b1;
#pragma unroll
    for (int p = 0; p < 8; ++p) {
      const int ke = half * 16 + p * 2;                     // B layout K index
      float2 t0 = *(const float2*)(Bt + ke);
      float2 t1 = *(const float2*)(Bt + 32 + ke);
      b0[2 * p] = (__bf16)t0.x;  b0[2 * p + 1] = (__bf16)t0.y;
      b1[2 * p] = (__bf16)t1.x;  b1[2 * p + 1] = (__bf16)t1.y;
    }
    v8f c = {};
    c = __builtin_amdgcn_wmma_f32_16x16x32_bf16(false, a0, false, b0, (short)0, c, false, false);
    c = __builtin_amdgcn_wmma_f32_16x16x32_bf16(false, a1, false, b1, (short)0, c, false, false);
#pragma unroll
    for (int r = 0; r < 8; ++r) {
      const int M = r + half * 8;
      const bool mv = (mbase[(size_t)M * S + n * 16 + row] != 0);
      mbits[r] |= (mv ? 1u : 0u) << n;
      float s = mv ? c[r] * SCALE2 : NEG_INF2;              // base-2 logits
      const float nm = fmaxf(lmax[r], s);
      lsum[r] = lsum[r] * __builtin_amdgcn_exp2f(lmax[r] - nm)
              + __builtin_amdgcn_exp2f(s - nm);
      lmax[r] = nm;
    }
  }

  // ---- one cross-lane combine per row (16-lane halves) ----
  float rmax[8], rinv[8];
#pragma unroll
  for (int r = 0; r < 8; ++r) {
    const float nm = red_max16(lmax[r]);
    const float ls = lsum[r] * __builtin_amdgcn_exp2f(lmax[r] - nm);
    const float rs = red_sum16(ls);
    rmax[r] = nm;
    rinv[r] = 1.0f / rs;
  }

  // ================= pass 2: recompute, normalize, stream probs (+ P*V) ===
  v8f acc[4] = {{}, {}, {}, {}};     // output d-tiles (wave 0)
  for (int n = 0; n < 32; ++n) {
    const float* Bt = Bbase + (size_t)(n * 16 + row) * Dh;
    if (n + 1 < 32)
      __builtin_prefetch(Bbase + (size_t)((n + 1) * 16 + row) * Dh, 0, 0);
    v16bf b0, b1;
#pragma unroll
    for (int p = 0; p < 8; ++p) {
      const int ke = half * 16 + p * 2;
      float2 t0 = *(const float2*)(Bt + ke);
      float2 t1 = *(const float2*)(Bt + 32 + ke);
      b0[2 * p] = (__bf16)t0.x;  b0[2 * p + 1] = (__bf16)t0.y;
      b1[2 * p] = (__bf16)t1.x;  b1[2 * p + 1] = (__bf16)t1.y;
    }
    v8f c = {};
    c = __builtin_amdgcn_wmma_f32_16x16x32_bf16(false, a0, false, b0, (short)0, c, false, false);
    c = __builtin_amdgcn_wmma_f32_16x16x32_bf16(false, a1, false, b1, (short)0, c, false, false);

    float pr[8];
#pragma unroll
    for (int r = 0; r < 8; ++r) {
      const int M = r + half * 8;
      const bool mv = (mbits[r] >> n) & 1u;                 // cached mask bit
      const float s = mv ? c[r] * SCALE2 : NEG_INF2;
      pr[r] = __builtin_amdgcn_exp2f(s - rmax[r]) * rinv[r];
      // Non-temporal: 480MB of probs must not thrash L2 (inputs live there).
      __builtin_nontemporal_store(pr[r], &probs[(size_t)M * S + n * 16 + row]);
    }

    if (w == 0) {
      // stage normalized P tile (C layout -> row-major [16][32])
#pragma unroll
      for (int r = 0; r < 8; ++r)
        stage[(r + half * 8) * 32 + (n & 1) * 16 + row] = pr[r];

      if (n & 1) {  // have a full 16x32 P slab -> one K=32 step of P*V
        v16bf pa;
#pragma unroll
        for (int p = 0; p < 8; ++p) {
          const int ke = half * 16 + p * 2;
          pa[2 * p]     = (__bf16)stage[row * 32 + ke];
          pa[2 * p + 1] = (__bf16)stage[row * 32 + ke + 1];
        }
        const int kb = (n >> 1) * 32;
#pragma unroll
        for (int nt = 0; nt < 4; ++nt) {
          v16bf vb;
#pragma unroll
          for (int p = 0; p < 8; ++p) {
            const int ke = half * 16 + p * 2;
            vb[2 * p]     = (__bf16)Vbase[(size_t)(kb + ke)     * Dh + nt * 16 + row];
            vb[2 * p + 1] = (__bf16)Vbase[(size_t)(kb + ke + 1) * Dh + nt * 16 + row];
          }
          acc[nt] = __builtin_amdgcn_wmma_f32_16x16x32_bf16(false, pa, false, vb,
                                                            (short)0, acc[nt], false, false);
        }
      }
    }
  }

  if (w == 0) {
#pragma unroll
    for (int nt = 0; nt < 4; ++nt)
#pragma unroll
      for (int r = 0; r < 8; ++r)
        __builtin_nontemporal_store(acc[nt][r],
            &obase[(size_t)(r + half * 8) * Dh + nt * 16 + row]);
  }
}

extern "C" void kernel_launch(void* const* d_in, const int* in_sizes, int n_in,
                              void* d_out, int out_size, void* d_ws, size_t ws_size,
                              hipStream_t stream) {
  (void)in_sizes; (void)n_in; (void)out_size; (void)d_ws; (void)ws_size;
  const float* q    = (const float*)d_in[0];
  const float* k    = (const float*)d_in[1];
  const float* v    = (const float*)d_in[2];
  const int*   mask = (const int*)  d_in[3];
  const float* o1   = (const float*)d_in[4];
  const float* o2   = (const float*)d_in[5];
  const float* p1   = (const float*)d_in[6];
  const float* p2   = (const float*)d_in[7];
  float* out = (float*)d_out;

  dim3 grid(512 / 16, 12, 8);   // (qtile, h, b)
  dim3 block(160);              // 5 wave32s: one per score matrix
  attn_multi_softmax_kernel<<<grid, block, 0, stream>>>(q, k, v, mask, o1, o2, p1, p2, out);
}
